// MultiScaleAutoencoderVQ_9320079032390
// MI455X (gfx1250) — matrix operand order repaired and checked by reading, not compiled, not run on hardware
//
#include <hip/hip_runtime.h>
#include <stdint.h>

// ---------------------------------------------------------------------------
// MI455X (gfx1250) multi-scale VQ autoencoder forward.
// Heavy convs: implicit GEMM, bf16 NHWC activations, v_wmma_f32_16x16x32_bf16,
// f32 accum, 2 N-tiles (32 out-channels) per wave for A-fragment reuse.
// VQ nearest-code search: WMMA (flat x E^T), LDS-staged A tile, 2 code tiles
// per iteration. Decoder nearest-upsample folded into the implicit-GEMM gather.
// ---------------------------------------------------------------------------

typedef __attribute__((ext_vector_type(16))) __bf16 v16bf;
typedef __attribute__((ext_vector_type(8)))  float  v8f;

union FragBF { uint4 q[2]; v16bf v; };

__device__ __forceinline__ unsigned short f2bf(float f) {
  unsigned int u = __float_as_uint(f);
  unsigned int r = (u + 0x7FFFu + ((u >> 16) & 1u)) >> 16;  // RNE
  return (unsigned short)r;
}

// ---- weight repack: OIHW f32 -> [OPad][taps][I] bf16 (zero-padded O) -------
__global__ void k_convert_w(const float* __restrict__ w, unsigned short* __restrict__ o,
                            int O, int OPad, int I, int taps) {
  long idx = (long)blockIdx.x * blockDim.x + threadIdx.x;
  long n = (long)OPad * taps * I;
  if (idx >= n) return;
  int co  = (int)(idx / ((long)taps * I));
  int r   = (int)(idx % ((long)taps * I));
  int tap = r / I, ci = r % I;
  float v = 0.f;
  if (co < O) v = w[((long)co * I + ci) * taps + tap];
  o[idx] = f2bf(v);
}

// ---- codebook -> bf16 + ||E||^2 -------------------------------------------
__global__ void k_convert_cb(const float* __restrict__ E, unsigned short* __restrict__ Eb,
                             float* __restrict__ esq) {
  int c = blockIdx.x * blockDim.x + threadIdx.x;
  if (c >= 4096) return;
  float s = 0.f;
  for (int j = 0; j < 32; ++j) {
    float v = E[c * 32 + j];
    s += v * v;
    Eb[c * 32 + j] = f2bf(v);
  }
  esq[c] = s;
}

// ---- first conv (Cin=3): direct VALU, f32 NCHW in -> bf16 NHWC out, relu ---
__global__ void k_conv_e0(const float* __restrict__ x, const float* __restrict__ w,
                          const float* __restrict__ bias, unsigned short* __restrict__ out) {
  long idx = (long)blockIdx.x * blockDim.x + threadIdx.x;
  const long total = 8L * 128 * 128 * 128;
  if (idx >= total) return;
  int co = (int)(idx & 127);
  long m = idx >> 7;
  int ox = (int)(m % 128); m /= 128;
  int oy = (int)(m % 128);
  int b  = (int)(m / 128);
  float acc = bias[co];
  for (int ci = 0; ci < 3; ++ci)
    for (int kh = 0; kh < 3; ++kh) {
      int iy = oy * 2 - 1 + kh;
      if (iy < 0 || iy >= 256) continue;
      for (int kw = 0; kw < 3; ++kw) {
        int ix = ox * 2 - 1 + kw;
        if (ix < 0 || ix >= 256) continue;
        acc += x[(((long)b * 3 + ci) * 256 + iy) * 256 + ix] *
               w[((co * 3 + ci) * 3 + kh) * 3 + kw];
      }
    }
  acc = fmaxf(acc, 0.f);
  out[idx] = f2bf(acc);
}

// ---- generic implicit-GEMM conv with WMMA bf16, optional 2x nearest-up -----
// in : bf16 NHWC (B,Hs,Ws,Cin)   wt : bf16 [CoutPad][ksz*ksz][Cin]
// Each wave: one 16-pixel M tile x 32 output channels (2 accumulators).
// outmode: 0 = bf16 NHWC, 1 = f32 NHWC, 2 = f32 NCHW (final 3-ch conv)
__global__ void k_conv_wmma(const unsigned short* __restrict__ in,
                            const unsigned short* __restrict__ wt,
                            const float* __restrict__ bias,
                            unsigned short* __restrict__ out_bf,
                            float* __restrict__ out_f,
                            int B, int Cin, int Cout, int CoutPad,
                            int Hs, int Ws, int up, int Hout, int Wout,
                            int stride, int pad, int ksz, int relu, int outmode) {
  const int lane = threadIdx.x & 31;
  const int wave = threadIdx.x >> 5;
  const int hh  = lane >> 4;     // wave32 half: selects K sub-range of fragment
  const int lid = lane & 15;
  const int ush = up >> 1;       // up==1 -> 0, up==2 -> 1
  const int Hin = Hs * up, Win = Ws * up;
  const long HW = (long)Hout * Wout;
  const long Mtot = (long)B * HW;
  const int  tilesN = (CoutPad + 31) >> 5;   // N-tile pairs
  const long tilesM = (Mtot + 15) >> 4;
  long tile = (long)blockIdx.x * (blockDim.x >> 5) + wave;
  if (tile >= tilesM * tilesN) return;
  long mtile = tile / tilesN;
  int  ntile = (int)(tile % tilesN);

  long mA = mtile * 16 + lid; if (mA >= Mtot) mA = Mtot - 1;   // A row per lane
  int bA = (int)(mA / HW);
  long rm = mA % HW;
  int oy = (int)(rm / Wout), ox = (int)(rm % Wout);
  const int n0 = ntile * 32 + lid;            // B columns for this lane
  const int n1 = n0 + 16;
  const bool hasN1 = (ntile * 32 + 16) < CoutPad;   // wave-uniform

  const int taps = ksz * ksz;
  const int aoff1 = hh ? 16 : 0;   // A k order used here: half0 K{0..15}
  const int aoff2 = hh ? 24 : 8;   //                  half1 K{16..31}
  const int woff  = hh ? 16 : 0;   // B same k order: half1 covers K 16..31
  v8f acc0 = {0.f,0.f,0.f,0.f,0.f,0.f,0.f,0.f};
  v8f acc1 = {0.f,0.f,0.f,0.f,0.f,0.f,0.f,0.f};

  int tap = 0;
  for (int kh = 0; kh < ksz; ++kh) {
    const int iy = oy * stride - pad + kh;
    const bool rowok = (iy >= 0) && (iy < Hin);
    const int sy = iy >> ush;     // folds nearest-upsample into the gather
    for (int kw = 0; kw < ksz; ++kw, ++tap) {
      const int ix = ox * stride - pad + kw;
      const bool ok = rowok && (ix >= 0) && (ix < Win);
      const int sx = ix >> ush;
      const unsigned short* pa = in + (((long)bA * Hs + sy) * Ws + sx) * Cin;
      const unsigned short* pw0 = wt + ((long)n0 * taps + tap) * Cin + woff;
      const unsigned short* pw1 = wt + ((long)n1 * taps + tap) * Cin + woff;
      for (int ci0 = 0; ci0 < Cin; ci0 += 32) {
        FragBF a;
        if (ok) {
          a.q[0] = *(const uint4*)(pa + ci0 + aoff1);
          a.q[1] = *(const uint4*)(pa + ci0 + aoff2);
        } else {
          a.q[0] = make_uint4(0u, 0u, 0u, 0u);
          a.q[1] = make_uint4(0u, 0u, 0u, 0u);
        }
        FragBF b0;
        b0.q[0] = *(const uint4*)(pw0 + ci0);
        b0.q[1] = *(const uint4*)(pw0 + ci0 + 8);
        __builtin_prefetch(pw0 + ci0 + Cin, 0, 1);  // global_prefetch_b8
        acc0 = __builtin_amdgcn_wmma_f32_16x16x32_bf16(false, a.v, false, b0.v,
                                                       (short)0, acc0, false, false);
        if (hasN1) {   // uniform branch: EXEC stays all-ones around WMMA
          FragBF b1;
          b1.q[0] = *(const uint4*)(pw1 + ci0);
          b1.q[1] = *(const uint4*)(pw1 + ci0 + 8);
          acc1 = __builtin_amdgcn_wmma_f32_16x16x32_bf16(false, a.v, false, b1.v,
                                                         (short)0, acc1, false, false);
        }
      }
    }
  }

#pragma unroll
  for (int t = 0; t < 2; ++t) {
    if (t == 1 && !hasN1) break;
    const v8f acc = t ? acc1 : acc0;
    const int n = t ? n1 : n0;
    const float bv = (n < Cout) ? bias[n] : 0.f;
#pragma unroll
    for (int r = 0; r < 8; ++r) {
      long m = mtile * 16 + r + (hh ? 8 : 0);   // D layout: half picks M rows 8..15
      if (m >= Mtot || n >= Cout) continue;
      float v = acc[r] + bv;
      if (relu) v = fmaxf(v, 0.f);
      if (outmode == 0) {
        out_bf[m * Cout + n] = f2bf(v);
      } else if (outmode == 1) {
        out_f[m * Cout + n] = v;
      } else {
        int b2 = (int)(m / HW); long r2 = m % HW;
        int y2 = (int)(r2 / Wout), x2 = (int)(r2 % Wout);
        out_f[(((long)b2 * Cout + n) * Hout + y2) * Wout + x2] = v;
      }
    }
  }
}

// ---- bilinear resize, f32 NHWC (half-pixel centers) ------------------------
__global__ void k_resize_bilinear(const float* __restrict__ in, float* __restrict__ out,
                                  int B, int Hin, int Win, int Hout, int Wout, int C) {
  long idx = (long)blockIdx.x * blockDim.x + threadIdx.x;
  long total = (long)B * Hout * Wout * C;
  if (idx >= total) return;
  int c = (int)(idx % C); long t = idx / C;
  int x = (int)(t % Wout); t /= Wout;
  int y = (int)(t % Hout);
  int b = (int)(t / Hout);
  float sy = ((float)y + 0.5f) * (float)Hin / (float)Hout - 0.5f;
  float sx = ((float)x + 0.5f) * (float)Win / (float)Wout - 0.5f;
  sy = fminf(fmaxf(sy, 0.f), (float)(Hin - 1));
  sx = fminf(fmaxf(sx, 0.f), (float)(Win - 1));
  int y0 = (int)sy, x0 = (int)sx;
  int y1 = min(y0 + 1, Hin - 1), x1 = min(x0 + 1, Win - 1);
  float fy = sy - (float)y0, fx = sx - (float)x0;
  const float* p = in + (long)b * Hin * Win * C + c;
  float v00 = p[((long)y0 * Win + x0) * C];
  float v01 = p[((long)y0 * Win + x1) * C];
  float v10 = p[((long)y1 * Win + x0) * C];
  float v11 = p[((long)y1 * Win + x1) * C];
  out[idx] = (v00 * (1.f - fx) + v01 * fx) * (1.f - fy) +
             (v10 * (1.f - fx) + v11 * fx) * fy;
}

// ---- VQ nearest code: 16-point tile, WMMA over 4096 codes, LDS-staged A ----
__global__ void k_vq_argmin(const float* __restrict__ flat,
                            const unsigned short* __restrict__ Eb,
                            const float* __restrict__ esq,
                            int npts, int* __restrict__ idx_out,
                            float* __restrict__ idxf_out) {
  __shared__ __align__(16) unsigned short sA[16 * 32];
  const int lane  = threadIdx.x;   // blockDim == 32, one wave
  const int mtile = blockIdx.x;
  {
    int ml  = lane >> 1;
    int ch0 = (lane & 1) * 16;
    long mg = (long)mtile * 16 + ml;
    if (mg >= npts) mg = npts - 1;
    const float* src = flat + mg * 32 + ch0;
    for (int j = 0; j < 16; ++j)
      sA[ml * 32 + ch0 + j] = f2bf(src[j]);
  }
  __syncthreads();
  const int hh = lane >> 4, lid = lane & 15;
  FragBF a;
  {
    const unsigned short* p = sA + lid * 32;
    a.q[0] = *(const uint4*)(p + (hh ? 16 : 0));
    a.q[1] = *(const uint4*)(p + (hh ? 24 : 8));
  }
  float bestv[8]; int besti[8];
#pragma unroll
  for (int r = 0; r < 8; ++r) { bestv[r] = 3.4e38f; besti[r] = 0; }

  for (int t = 0; t < 256; t += 2) {   // 2 code tiles per iter, A reused
    const int code0 = t * 16 + lid;
    const int code1 = code0 + 16;
    FragBF b0, b1;
    const unsigned short* pw0 = Eb + (long)code0 * 32 + (hh ? 16 : 0);
    b0.q[0] = *(const uint4*)(pw0);
    b0.q[1] = *(const uint4*)(pw0 + 8);
    b1.q[0] = *(const uint4*)(pw0 + 32);
    b1.q[1] = *(const uint4*)(pw0 + 40);
    v8f c0 = {0.f,0.f,0.f,0.f,0.f,0.f,0.f,0.f};
    v8f c1 = {0.f,0.f,0.f,0.f,0.f,0.f,0.f,0.f};
    c0 = __builtin_amdgcn_wmma_f32_16x16x32_bf16(false, a.v, false, b0.v,
                                                 (short)0, c0, false, false);
    c1 = __builtin_amdgcn_wmma_f32_16x16x32_bf16(false, a.v, false, b1.v,
                                                 (short)0, c1, false, false);
    const float en0 = esq[code0];
    const float en1 = esq[code1];
#pragma unroll
    for (int r = 0; r < 8; ++r) {
      float d0 = en0 - 2.f * c0[r];  // ||flat||^2 constant per row: drop for argmin
      float d1 = en1 - 2.f * c1[r];
      if (d0 < bestv[r] || (d0 == bestv[r] && code0 < besti[r])) {
        bestv[r] = d0; besti[r] = code0;
      }
      if (d1 < bestv[r] || (d1 == bestv[r] && code1 < besti[r])) {
        bestv[r] = d1; besti[r] = code1;
      }
    }
  }
  // min-reduce across the 16 lanes of each half (lane bit4 untouched)
#pragma unroll
  for (int r = 0; r < 8; ++r) {
    for (int s = 1; s < 16; s <<= 1) {
      float ov = __shfl_xor(bestv[r], s, 32);
      int   oi = __shfl_xor(besti[r], s, 32);
      if (ov < bestv[r] || (ov == bestv[r] && oi < besti[r])) {
        bestv[r] = ov; besti[r] = oi;
      }
    }
  }
  if (lid == 0) {
#pragma unroll
    for (int r = 0; r < 8; ++r) {
      long m = (long)mtile * 16 + r + (hh ? 8 : 0);
      if (m < npts) { idx_out[m] = besti[r]; idxf_out[m] = (float)besti[r]; }
    }
  }
}

// ---- small elementwise / reduction helpers --------------------------------
__global__ void k_gather_zq(const float* __restrict__ E, const int* __restrict__ idx,
                            float* __restrict__ zq, long n) {
  long i = (long)blockIdx.x * blockDim.x + threadIdx.x;
  if (i >= n) return;
  int c = (int)(i & 31);
  long m = i >> 5;
  zq[i] = E[(long)idx[m] * 32 + c];
}

__global__ void k_loss_partial(const float* __restrict__ a, const float* __restrict__ b,
                               float* __restrict__ part, long n) {
  __shared__ float sm[256];
  long i = (long)blockIdx.x * 256 + threadIdx.x;
  float d = 0.f;
  if (i < n) { float t = a[i] - b[i]; d = t * t; }
  sm[threadIdx.x] = d;
  __syncthreads();
  for (int s = 128; s > 0; s >>= 1) {
    if (threadIdx.x < s) sm[threadIdx.x] += sm[threadIdx.x + s];
    __syncthreads();
  }
  if (threadIdx.x == 0) part[blockIdx.x] = sm[0];
}

__global__ void k_loss_final(const float* __restrict__ part, int nb, float scale,
                             float* __restrict__ loss) {
  if (threadIdx.x == 0 && blockIdx.x == 0) {
    float s = 0.f;
    for (int i = 0; i < nb; ++i) s += part[i];
    loss[0] += s * scale;
  }
}

__global__ void k_sub(float* __restrict__ out, const float* __restrict__ a,
                      const float* __restrict__ b, long n) {
  long i = (long)blockIdx.x * blockDim.x + threadIdx.x;
  if (i < n) out[i] = a[i] - b[i];
}

__global__ void k_f2bf_arr(const float* __restrict__ in, unsigned short* __restrict__ out,
                           long n) {
  long i = (long)blockIdx.x * blockDim.x + threadIdx.x;
  if (i < n) out[i] = f2bf(in[i]);
}

__global__ void k_zero1(float* __restrict__ p) {
  if (threadIdx.x == 0 && blockIdx.x == 0) p[0] = 0.f;
}

// ---------------------------------------------------------------------------
extern "C" void kernel_launch(void* const* d_in, const int* in_sizes, int n_in,
                              void* d_out, int out_size, void* d_ws, size_t ws_size,
                              hipStream_t stream) {
  (void)in_sizes; (void)n_in; (void)out_size; (void)ws_size;
  const float* X     = (const float*)d_in[0];
  const float* W_E0  = (const float*)d_in[1];  const float* B_E0  = (const float*)d_in[2];
  const float* W_E1f = (const float*)d_in[3];  const float* B_E1  = (const float*)d_in[4];
  const float* W_E2f = (const float*)d_in[5];  const float* B_E2  = (const float*)d_in[6];
  const float* W_E3f = (const float*)d_in[7];  const float* B_E3  = (const float*)d_in[8];
  const float* W_Qf  = (const float*)d_in[9];  const float* B_Q   = (const float*)d_in[10];
  const float* CB    = (const float*)d_in[11];
  const float* W_RESf= (const float*)d_in[12]; const float* B_RES = (const float*)d_in[13];
  const float* W_D0f = (const float*)d_in[14]; const float* B_D0  = (const float*)d_in[15];
  const float* W_U1f = (const float*)d_in[16]; const float* B_U1  = (const float*)d_in[17];
  const float* W_U2f = (const float*)d_in[18]; const float* B_U2  = (const float*)d_in[19];
  const float* W_U3f = (const float*)d_in[20]; const float* B_U3  = (const float*)d_in[21];
  const float* W_U4f = (const float*)d_in[22]; const float* B_U4  = (const float*)d_in[23];
  const float* W_OUTf= (const float*)d_in[24]; const float* B_OUT = (const float*)d_in[25];

  char* ws = (char*)d_ws;
  size_t off = 0;
  auto balloc = [&](size_t bytes) -> void* {
    off = (off + 255) & ~(size_t)255;
    void* p = (void*)(ws + off);
    off += bytes;
    return p;
  };
  auto aus = [&](size_t n) { return (unsigned short*)balloc(n * 2); };
  auto af  = [&](size_t n) { return (float*)balloc(n * 4); };

  // weight buffers (bf16, GEMM layout)
  unsigned short* wE1 = aus(256ULL * 9 * 128);
  unsigned short* wE2 = aus(256ULL * 9 * 256);
  unsigned short* wE3 = aus(512ULL * 9 * 256);
  unsigned short* wQ  = aus(32ULL * 512);
  unsigned short* wRS = aus(320ULL * 9 * 32);
  unsigned short* wD0 = aus(512ULL * 32);
  unsigned short* wU1 = aus(256ULL * 9 * 512);
  unsigned short* wU2 = aus(256ULL * 9 * 256);
  unsigned short* wU3 = aus(128ULL * 9 * 256);
  unsigned short* wU4 = aus(128ULL * 9 * 128);
  unsigned short* wOT = aus(16ULL * 9 * 128);
  unsigned short* Ebf = aus(4096ULL * 32);
  float* esq = af(4096);
  // activations (bf16 NHWC)
  unsigned short* aE0 = aus(8ULL * 128 * 128 * 128);
  unsigned short* aE1 = aus(8ULL * 64 * 64 * 256);
  unsigned short* aE2 = aus(8ULL * 32 * 32 * 256);
  unsigned short* aE3 = aus(8ULL * 16 * 16 * 512);
  unsigned short* aD0 = aus(8ULL * 16 * 16 * 512);
  unsigned short* aU1 = aus(8ULL * 32 * 32 * 256);
  unsigned short* aU2 = aus(8ULL * 64 * 64 * 256);
  unsigned short* aU3 = aus(8ULL * 128 * 128 * 128);
  unsigned short* aU4 = aus(8ULL * 256 * 256 * 128);
  unsigned short* fbf = aus(65536);
  unsigned short* zbf = aus(65536);
  // f32 VQ buffers (NHWC, B=8, 16x16x32 at full scale)
  float* zf   = af(65536);
  float* resf = af(65536);
  float* xif  = af(65536);
  float* zqf  = af(65536);
  float* zupf = af(65536);
  float* zmapf= af(65536);
  float* ff   = af(65536);
  float* part = af(1024);
  int*   idxi = (int*)balloc(2048 * 4);

  float* outf = (float*)d_out;
  float* loss_ptr = outf + 1572864 + 5440;

  auto g1 = [](long n) { return dim3((unsigned)((n + 255) / 256)); };
  auto convw = [&](const float* w, unsigned short* o, int O, int OPad, int I, int taps) {
    long n = (long)OPad * taps * I;
    k_convert_w<<<g1(n), dim3(256), 0, stream>>>(w, o, O, OPad, I, taps);
  };
  auto conv = [&](const unsigned short* in, const unsigned short* wt, const float* bias,
                  unsigned short* obf, float* of,
                  int B, int Cin, int Cout, int CoutPad, int Hs, int Ws, int up,
                  int Hout, int Wout, int stride, int pad, int ksz, int relu, int outmode) {
    long Mtot = (long)B * Hout * Wout;
    long tiles = ((Mtot + 15) / 16) * ((CoutPad + 31) / 32);
    long blocks = (tiles + 7) / 8;   // 8 waves per 256-thread block
    k_conv_wmma<<<dim3((unsigned)blocks), dim3(256), 0, stream>>>(
        in, wt, bias, obf, of, B, Cin, Cout, CoutPad, Hs, Ws, up,
        Hout, Wout, stride, pad, ksz, relu, outmode);
  };

  // ---- weight / codebook repack ----
  convw(W_E1f, wE1, 256, 256, 128, 9);
  convw(W_E2f, wE2, 256, 256, 256, 9);
  convw(W_E3f, wE3, 512, 512, 256, 9);
  convw(W_Qf,  wQ,   32,  32, 512, 1);
  convw(W_RESf,wRS, 320, 320,  32, 9);   // 10 convs packed contiguously
  convw(W_D0f, wD0, 512, 512,  32, 1);
  convw(W_U1f, wU1, 256, 256, 512, 9);
  convw(W_U2f, wU2, 256, 256, 256, 9);
  convw(W_U3f, wU3, 128, 128, 256, 9);
  convw(W_U4f, wU4, 128, 128, 128, 9);
  convw(W_OUTf,wOT,   3,  16, 128, 9);
  k_convert_cb<<<dim3(16), dim3(256), 0, stream>>>(CB, Ebf, esq);

  // ---- encoder ----
  k_conv_e0<<<g1(8L * 128 * 128 * 128), dim3(256), 0, stream>>>(X, W_E0, B_E0, aE0);
  conv(aE0, wE1, B_E1, aE1, nullptr, 8, 128, 256, 256, 128, 128, 1, 64, 64, 2, 1, 3, 1, 0);
  conv(aE1, wE2, B_E2, aE2, nullptr, 8, 256, 256, 256,  64,  64, 1, 32, 32, 2, 1, 3, 1, 0);
  conv(aE2, wE3, B_E3, aE3, nullptr, 8, 256, 512, 512,  32,  32, 1, 16, 16, 2, 1, 3, 1, 0);
  conv(aE3, wQ,  B_Q,  nullptr, zf,  8, 512,  32,  32,  16,  16, 1, 16, 16, 1, 0, 1, 0, 1);

  hipMemcpyAsync(resf, zf, 65536 * sizeof(float), hipMemcpyDeviceToDevice, stream);
  k_zero1<<<dim3(1), dim3(32), 0, stream>>>(loss_ptr);

  // ---- multi-scale residual VQ ----
  const int STEPS[10] = {1, 2, 3, 4, 5, 6, 8, 10, 13, 16};
  long idxoff = 1572864;
  for (int i = 0; i < 10; ++i) {
    int s = STEPS[i];
    int npts = 8 * s * s;
    long nel = (long)npts * 32;
    k_resize_bilinear<<<g1(nel), dim3(256), 0, stream>>>(resf, xif, 8, 16, 16, s, s, 32);
    k_vq_argmin<<<dim3((unsigned)((npts + 15) / 16)), dim3(32), 0, stream>>>(
        xif, Ebf, esq, npts, idxi, outf + idxoff);
    idxoff += npts;
    k_gather_zq<<<g1(nel), dim3(256), 0, stream>>>(CB, idxi, zqf, nel);
    int nb = (int)((nel + 255) / 256);
    k_loss_partial<<<dim3(nb), dim3(256), 0, stream>>>(zqf, xif, part, nel);
    k_loss_final<<<dim3(1), dim3(32), 0, stream>>>(part, nb, 1.25f / (float)nel, loss_ptr);
    // zst == zq in forward; resize back to 16x16 and run this scale's conv
    k_resize_bilinear<<<g1(65536), dim3(256), 0, stream>>>(zqf, zupf, 8, s, s, 16, 16, 32);
    k_f2bf_arr<<<g1(65536), dim3(256), 0, stream>>>(zupf, zbf, 65536);
    conv(zbf, wRS + (size_t)i * 32 * 9 * 32, B_RES + i * 32, nullptr, zmapf,
         8, 32, 32, 32, 16, 16, 1, 16, 16, 1, 1, 3, 0, 1);
    k_sub<<<g1(65536), dim3(256), 0, stream>>>(resf, resf, zmapf, 65536);
  }

  // ---- decoder: f = sum_k zmap_k = z - res_final ----
  k_sub<<<g1(65536), dim3(256), 0, stream>>>(ff, zf, resf, 65536);
  k_f2bf_arr<<<g1(65536), dim3(256), 0, stream>>>(ff, fbf, 65536);
  conv(fbf, wD0, B_D0, aD0, nullptr, 8,  32, 512, 512,  16,  16, 1,  16,  16, 1, 0, 1, 1, 0);
  conv(aD0, wU1, B_U1, aU1, nullptr, 8, 512, 256, 256,  16,  16, 2,  32,  32, 1, 1, 3, 1, 0);
  conv(aU1, wU2, B_U2, aU2, nullptr, 8, 256, 256, 256,  32,  32, 2,  64,  64, 1, 1, 3, 1, 0);
  conv(aU2, wU3, B_U3, aU3, nullptr, 8, 256, 128, 128,  64,  64, 2, 128, 128, 1, 1, 3, 1, 0);
  conv(aU3, wU4, B_U4, aU4, nullptr, 8, 128, 128, 128, 128, 128, 2, 256, 256, 1, 1, 3, 1, 0);
  conv(aU4, wOT, B_OUT, nullptr, outf, 8, 128, 3, 16, 256, 256, 1, 256, 256, 1, 1, 3, 0, 2);
}
